// MHSG_20452634264254
// MI455X (gfx1250) — matrix operand order repaired and checked by reading, not compiled, not run on hardware
//
#include <hip/hip_runtime.h>
#include <hip/hip_bf16.h>
#include <math.h>

typedef __attribute__((ext_vector_type(16))) _Float16 v16h;
typedef __attribute__((ext_vector_type(8)))  float    v8f;

#define ALPH      0.8f
#define BSZ       64
#define CSZ       64
#define NSZ       500
#define TSZ       48
#define ROW       24000     // NSZ*TSZ
#define NPAD      512
#define INV_SQRTC 0.125f    // 1/sqrt(64)

// ---------------------------------------------------------------------------
// k0: s_rowsum[n] = sum_m s[n,m]   (500 blocks x 128 threads)
// ---------------------------------------------------------------------------
__global__ void k0_srowsum(const float* __restrict__ s, float* __restrict__ srs) {
    __shared__ float red[128];
    const int n = blockIdx.x;
    float acc = 0.f;
    for (int j = threadIdx.x; j < NSZ; j += 128) acc += s[n * NSZ + j];
    red[threadIdx.x] = acc; __syncthreads();
    for (int off = 64; off > 0; off >>= 1) {
        if (threadIdx.x < off) red[threadIdx.x] += red[threadIdx.x + off];
        __syncthreads();
    }
    if (threadIdx.x == 0) srs[n] = red[0];
}

// ---------------------------------------------------------------------------
// k1: one block per (b,c).  Single HBM pass over x: stage relu-scaled row in
// LDS (96 KB), block-reduce max, per-n exp sums, normalize -> xws[bid*500+n].
// ---------------------------------------------------------------------------
__global__ void k1_softmax_rowsum(const float* __restrict__ x,
                                  const float* __restrict__ srs,
                                  float* __restrict__ xws) {
    extern __shared__ float buf[];      // ROW floats (dynamic, 96000 B)
    __shared__ float bins[NSZ];
    __shared__ float red[256];
    const int tid = threadIdx.x;
    const long long base = (long long)blockIdx.x * ROW;

    // Pass A: compute v = relu(x * rowsum / sqrt(C)), keep in LDS, track max.
    float lmax = 0.f;                   // relu output >= 0
    for (int k = tid; k < ROW; k += 256) {
        int i = k / NSZ;                // 0..47 (time block)
        int v = k - i * NSZ;            // 0..499 (node)
        float mult = (float)(TSZ - 1 - i) * ALPH + srs[v];
        float val  = x[base + k] * mult * INV_SQRTC;
        val = fmaxf(val, 0.f);
        buf[k] = val;
        lmax = fmaxf(lmax, val);
    }
    red[tid] = lmax; __syncthreads();
    for (int off = 128; off > 0; off >>= 1) {
        if (tid < off) red[tid] = fmaxf(red[tid], red[tid + off]);
        __syncthreads();
    }
    const float mx = red[0];
    __syncthreads();

    // Pass B: per-n sums of exp(v - mx) over the T=48 contiguous elements.
    float zloc = 0.f;
    for (int n = tid; n < NSZ; n += 256) {
        float s = 0.f;
        const int kb = n * TSZ;
        #pragma unroll 8
        for (int t = 0; t < TSZ; ++t) s += __expf(buf[kb + t] - mx);
        bins[n] = s;
        zloc += s;
    }
    red[tid] = zloc; __syncthreads();
    for (int off = 128; off > 0; off >>= 1) {
        if (tid < off) red[tid] += red[tid + off];
        __syncthreads();
    }
    const float invZ = 1.f / red[0];
    for (int n = tid; n < NSZ; n += 256)
        xws[(long long)blockIdx.x * NSZ + n] = bins[n] * invZ;
}

// ---------------------------------------------------------------------------
// k2: one block per (batch b, 16-row stripe).  Load X=x_w_s[b] (64x500 f32)
// transposed into LDS as f16 Xt[512][64] (zero padded), compute the 16x512
// gram stripe with v_wmma_f32_16x16x32_f16 (K=64 -> 2 k-steps), then
// row-softmax(relu(gram)/8) over the 500 valid columns and write out.
// ---------------------------------------------------------------------------
__global__ void k2_gram_softmax(const float* __restrict__ xws,
                                float* __restrict__ out) {
    extern __shared__ char smem[];
    _Float16* Xt = (_Float16*)smem;                               // [NPAD][64]
    float*    gs = (float*)(smem + NPAD * CSZ * sizeof(_Float16)); // [16][NPAD]

    const int tid  = threadIdx.x;
    const int b    = blockIdx.x >> 5;   // batch
    const int tile = blockIdx.x & 31;   // 16-row stripe index
    const int n0   = tile * 16;

    // Load + transpose + f32->f16 convert (pad columns n>=500 with 0).
    for (int idx = tid; idx < CSZ * NPAD; idx += 256) {
        int c = idx >> 9;               // /NPAD
        int n = idx & (NPAD - 1);
        float v = (n < NSZ) ? xws[((long long)b * CSZ + c) * NSZ + n] : 0.f;
        Xt[n * CSZ + c] = (_Float16)v;
    }
    __syncthreads();

    const int w  = tid >> 5;            // wave 0..7
    const int l  = tid & 31;            // lane
    const int hs = l >> 4;              // half-wave select
    const int ml = l & 15;

    // A fragments: rows n0..n0+15, K=c.  ISA f16 A layout:
    //   half j<8  -> K = 8*hs + j ; half j>=8 -> K = 16 + 8*hs + (j-8)
    v16h a0, a1;
    {
        const _Float16* ra = &Xt[(n0 + ml) * CSZ];
        #pragma unroll
        for (int j = 0; j < 8; ++j) {
            a0[j]     = ra[ 0 + 8 * hs + j];
            a0[j + 8] = ra[16 + 8 * hs + j];
            a1[j]     = ra[32 + 8 * hs + j];
            a1[j + 8] = ra[48 + 8 * hs + j];
        }
    }

    // 32 column tiles / 8 waves = 4 tiles per wave.
    #pragma unroll
    for (int i = 0; i < 4; ++i) {
        const int m0 = (w + i * 8) * 16;
        const _Float16* rb = &Xt[(m0 + ml) * CSZ];
        // B fragment: col = m0+ml, half j<8 -> K = 8*hs + j ; j>=8 -> K = 16 + 8*hs + (j-8)
        v16h b0, b1;
        #pragma unroll
        for (int j = 0; j < 16; ++j) {
            b0[j] = rb[ 0 + 8 * hs + j + (j >= 8 ? 8 : 0)];
            b1[j] = rb[32 + 8 * hs + j + (j >= 8 ? 8 : 0)];
        }
        v8f acc = {};
        acc = __builtin_amdgcn_wmma_f32_16x16x32_f16(
                  false, a0, false, b0, (short)0, acc, false, false);
        acc = __builtin_amdgcn_wmma_f32_16x16x32_f16(
                  false, a1, false, b1, (short)0, acc, false, false);
        // D layout: VGPR r -> row 8*hs + r, col ml
        #pragma unroll
        for (int r = 0; r < 8; ++r)
            gs[(8 * hs + r) * NPAD + m0 + ml] = acc[r];
    }
    __syncthreads();

    // Row softmax: wave w handles local rows 2w and 2w+1.
    #pragma unroll
    for (int rr = 0; rr < 2; ++rr) {
        const int r = w * 2 + rr;
        const int n = n0 + r;
        if (n >= NSZ) continue;         // uniform per wave
        const float* g = &gs[r * NPAD];
        float lmax = 0.f;               // relu logits >= 0
        for (int m = l; m < NSZ; m += 32)
            lmax = fmaxf(lmax, fmaxf(g[m], 0.f) * INV_SQRTC);
        for (int off = 16; off > 0; off >>= 1)
            lmax = fmaxf(lmax, __shfl_xor(lmax, off, 32));
        float zs = 0.f;
        for (int m = l; m < NSZ; m += 32)
            zs += __expf(fmaxf(g[m], 0.f) * INV_SQRTC - lmax);
        for (int off = 16; off > 0; off >>= 1)
            zs += __shfl_xor(zs, off, 32);
        const float invZ = 1.f / zs;
        float* orow = &out[((long long)b * NSZ + n) * NSZ];
        for (int m = l; m < NSZ; m += 32)
            orow[m] = __expf(fmaxf(g[m], 0.f) * INV_SQRTC - lmax) * invZ;
    }
}

// ---------------------------------------------------------------------------
extern "C" void kernel_launch(void* const* d_in, const int* in_sizes, int n_in,
                              void* d_out, int out_size, void* d_ws, size_t ws_size,
                              hipStream_t stream) {
    const float* x = (const float*)d_in[0];   // [64,64,500,48]
    const float* s = (const float*)d_in[1];   // [500,500]
    float* out = (float*)d_out;               // [64,500,500]

    float* srs = (float*)d_ws;                // 500 floats (padded to 512)
    float* xws = srs + 512;                   // 4096*500 floats (~8 MB)

    k0_srowsum<<<NSZ, 128, 0, stream>>>(s, srs);

    k1_softmax_rowsum<<<BSZ * CSZ, 256, ROW * sizeof(float), stream>>>(x, srs, xws);

    const size_t sm2 = NPAD * CSZ * sizeof(_Float16) + 16 * NPAD * sizeof(float);
    k2_gram_softmax<<<BSZ * 32, 256, sm2, stream>>>(xws, out);
}
